// RSSM_31327491457094
// MI455X (gfx1250) — matrix operand order repaired and checked by reading, not compiled, not run on hardware
//
#include <hip/hip_runtime.h>
#include <hip/hip_bf16.h>

typedef __attribute__((ext_vector_type(16))) _Float16 v16h;
typedef __attribute__((ext_vector_type(8)))  float    v8f;

#define NB   64      // batch
#define NT   64      // timesteps
#define DET  2048
#define HID  512
#define SDIM 1024    // STOCH*CLASSES
#define NBLK 8
#define DPER 256
#define KCHUNK 64

// ---- optional gfx1250 async global->LDS path (compile-safe probe) ----
#if defined(__has_builtin)
#if __has_builtin(__builtin_amdgcn_global_load_async_to_lds_b128) && \
    __has_builtin(__builtin_amdgcn_s_wait_asynccnt)
#define HAVE_ASYNC_LDS 1
#endif
#endif
#ifndef HAVE_ASYNC_LDS
#define HAVE_ASYNC_LDS 0
#endif

#if HAVE_ASYNC_LDS
typedef int v4i_ __attribute__((vector_size(4 * sizeof(int))));
typedef __attribute__((address_space(1))) v4i_ gv4i;   // global (AS1)
typedef __attribute__((address_space(3))) v4i_ lv4i;   // LDS (AS3)
#endif

__device__ __forceinline__ float sigm_(float x){ return 1.0f/(1.0f+expf(-x)); }
__device__ __forceinline__ float silu_(float x){ return x/(1.0f+expf(-x)); }

__device__ __forceinline__ float gumbel_hash_(unsigned int x){
  x ^= 2747636419u; x *= 2654435769u; x ^= x>>16; x *= 2654435769u; x ^= x>>16; x *= 2654435769u;
  float u = (float)((x >> 8) + 1u) * (1.0f/16777218.0f);
  return -logf(-logf(u));
}

// ---------------------------------------------------------------------------
// Fused GEMM:  C[M=64,N] = concat(A0,A1)[M,K](f16) @ BT^T(f16)  (+bias, +rms/silu)
// BT is f16 [N][K] (pre-transposed).  One workgroup = 16-row M tile x full N.
// 8 waves, NTW 16-wide N tiles per wave (N = 128*NTW).
// MODE 0: C(f32) = acc + bias.   MODE 1: C(f16) = silu(rms(acc+bias) * scale).
// blockIdx.y = block index for block-diagonal linears (offsets via a0_blk/c_blk).
// A tile double-buffered in LDS, filled by async global->LDS DMA when available.
// ---------------------------------------------------------------------------
template<int NTW, int MODE>
__global__ __launch_bounds__(256) void gemm_wmma(
    const _Float16* __restrict__ A0, int lda0, int len0,
    const _Float16* __restrict__ A1, int lda1, int len1,
    const _Float16* __restrict__ BT,
    const float* __restrict__ bias,
    const float* __restrict__ scale,
    void* __restrict__ Cv, long long ldc,
    int N, int K, int a0_blk, int c_blk)
{
  __shared__ _Float16 As[2][16*KCHUNK];
  __shared__ float s_rowsq[16];

  const int blk = blockIdx.y;
  A0   += (size_t)blk * a0_blk;
  BT   += (size_t)blk * (size_t)N * (size_t)K;
  bias += (size_t)blk * N;

  const int m0   = blockIdx.x * 16;
  const int tid  = threadIdx.x;
  const int lane = tid & 31;
  const int wave = tid >> 5;
  const int nlo  = lane & 15;   // N col within tile (B/C/D); also M row for A frag
  const int hi   = lane >> 4;

  v8f acc[NTW];
#pragma unroll
  for (int t = 0; t < NTW; ++t)
#pragma unroll
    for (int i = 0; i < 8; ++i) acc[t][i] = 0.0f;

#if HAVE_ASYNC_LDS
  // issue one 16x64-half chunk as 128 lanes x 16B async DMA
  auto issue = [&](int buf, int kc) {
    if (tid < 128) {
      int r  = tid >> 3;            // 0..15 rows
      int c8 = (tid & 7) * 8;       // 8 halves = 16B per lane
      int gc = kc + c8;
      const _Float16* gp = (gc < len0) ? (A0 + (size_t)(m0 + r)*lda0 + gc)
                                       : (A1 + (size_t)(m0 + r)*lda1 + (gc - len0));
      _Float16* lp = &As[buf][r*KCHUNK + c8];
      __builtin_amdgcn_global_load_async_to_lds_b128((gv4i*)gp, (lv4i*)lp, 0, 0);
    }
  };
  int buf = 0;
  issue(0, 0);
#endif

  for (int kc = 0; kc < K; kc += KCHUNK) {
#if HAVE_ASYNC_LDS
    bool more = (kc + KCHUNK) < K;
    if (more) issue(buf ^ 1, kc + KCHUNK);      // prefetch next chunk
    if (more) __builtin_amdgcn_s_wait_asynccnt(1);   // current chunk complete
    else      __builtin_amdgcn_s_wait_asynccnt(0);
    __syncthreads();
    const unsigned int* As32 = (const unsigned int*)As[buf];
#else
    for (int i = tid; i < 16*KCHUNK; i += 256) {
      int r = i / KCHUNK, c = i % KCHUNK;
      int gc = kc + c;
      As[0][i] = (gc < len0) ? A0[(size_t)(m0 + r)*lda0 + gc]
                             : A1[(size_t)(m0 + r)*lda1 + (gc - len0)];
    }
    __syncthreads();
    const unsigned int* As32 = (const unsigned int*)As[0];
#endif
#pragma unroll
    for (int kk = 0; kk < KCHUNK; kk += 32) {
      // A fragment: lane holds M=nlo; vgpr w: k = 16*(w>=4) + 2*(w&3) + h + 8*hi
      union { v16h v; unsigned int u[8]; } af;
      int abase = (nlo*KCHUNK + kk + 8*hi) >> 1;
#pragma unroll
      for (int w = 0; w < 4; ++w) { af.u[w] = As32[abase + w]; af.u[4+w] = As32[abase + 8 + w]; }
#pragma unroll
      for (int t = 0; t < NTW; ++t) {
        int n = (wave*NTW + t)*16 + nlo;
        int kbase = kc + kk + 8*hi;      // lanes 0-15: K=kk..kk+7, kk+16..kk+23 ; 16-31: +8
        union { v16h v; uint4 q[2]; } bf;
        const uint4* bp = (const uint4*)(BT + (size_t)n*K + kbase);
        bf.q[0] = bp[0]; bf.q[1] = bp[2];
        acc[t] = __builtin_amdgcn_wmma_f32_16x16x32_f16(
            false, af.v, false, bf.v, (short)0, acc[t], false, false);
      }
    }
    __syncthreads();
#if HAVE_ASYNC_LDS
    buf ^= 1;
#endif
  }

  if constexpr (MODE == 1) {
    _Float16* C = (_Float16*)Cv + (size_t)blk * c_blk;
    if (tid < 16) s_rowsq[tid] = 0.0f;
    __syncthreads();
#pragma unroll
    for (int t = 0; t < NTW; ++t) {
      int n = (wave*NTW + t)*16 + nlo;
      float b = bias[n];
#pragma unroll
      for (int i = 0; i < 8; ++i) {
        float v = acc[t][i] + b;
        atomicAdd(&s_rowsq[i + 8*hi], v*v);
      }
    }
    __syncthreads();
#pragma unroll
    for (int t = 0; t < NTW; ++t) {
      int n = (wave*NTW + t)*16 + nlo;
      float b = bias[n];
      float sn = scale[n];
#pragma unroll
      for (int i = 0; i < 8; ++i) {
        int row = i + 8*hi;
        float rs = rsqrtf(s_rowsq[row] / (float)N + 1e-8f);
        float y = (acc[t][i] + b) * rs * sn;
        C[(size_t)(m0 + row)*ldc + n] = (_Float16)silu_(y);
      }
    }
  } else {
    float* C = (float*)Cv + (size_t)blk * c_blk;
#pragma unroll
    for (int t = 0; t < NTW; ++t) {
      int n = (wave*NTW + t)*16 + nlo;
      float b = bias[n];
#pragma unroll
      for (int i = 0; i < 8; ++i)
        C[(size_t)(m0 + i + 8*hi)*ldc + n] = acc[t][i] + b;
    }
  }
}

// -------------------- elementwise / prep kernels ---------------------------

__global__ void transpose_f16_k(const float* __restrict__ src, _Float16* __restrict__ dst,
                                int nblk, int K, int N)
{
  int idx = blockIdx.x*blockDim.x + threadIdx.x;
  int per = K*N;
  if (idx >= nblk*per) return;
  int blk = idx / per, r = idx % per;
  int n = r / K, k = r % K;
  dst[idx] = (_Float16)src[(size_t)blk*per + (size_t)k*N + n];
}

__global__ void fill0f_k(float* p, int n){
  int i = blockIdx.x*blockDim.x + threadIdx.x;
  if (i < n) p[i] = 0.0f;
}
__global__ void fill0h_k(_Float16* p, int n){
  int i = blockIdx.x*blockDim.x + threadIdx.x;
  if (i < n) p[i] = (_Float16)0.0f;
}

// mask carries with is_first, normalize action, convert step inputs to f16
__global__ void begin_step_k(const float* __restrict__ deter, const _Float16* __restrict__ stoch,
                             const float* __restrict__ actions, const float* __restrict__ tokens,
                             const unsigned char* __restrict__ is_first, int t,
                             _Float16* __restrict__ deter_m, _Float16* __restrict__ stoch_m,
                             _Float16* __restrict__ act_n, _Float16* __restrict__ tok16)
{
  int idx = blockIdx.x*blockDim.x + threadIdx.x;
  if (idx < NB*DET) {
    int b = idx >> 11; float m = is_first[b*NT + t] ? 0.0f : 1.0f;
    deter_m[idx] = (_Float16)(deter[idx]*m);
  }
  if (idx < NB*SDIM) {
    int b = idx >> 10; float m = is_first[b*NT + t] ? 0.0f : 1.0f;
    stoch_m[idx] = (_Float16)((float)stoch[idx]*m);
  }
  if (idx < NB*HID) {
    int b = idx >> 9;
    tok16[idx] = (_Float16)tokens[((size_t)b*NT + t)*HID + (idx & 511)];
  }
  if (idx < NB*64) {
    int b = idx >> 6; float m = is_first[b*NT + t] ? 0.0f : 1.0f;
    float a = actions[((size_t)b*NT + t)*64 + (idx & 63)] * m;
    act_n[idx] = (_Float16)(a / fmaxf(fabsf(a), 1.0f));
  }
}

__global__ void rmsnorm_silu_rows_k(const float* __restrict__ X, const float* __restrict__ scale,
                                    _Float16* __restrict__ Y, int N)
{
  int row = blockIdx.x;
  const float* x = X + (size_t)row*N;
  __shared__ float red[256];
  float s = 0.0f;
  for (int j = threadIdx.x; j < N; j += 256) { float v = x[j]; s += v*v; }
  red[threadIdx.x] = s; __syncthreads();
  for (int o = 128; o > 0; o >>= 1) { if (threadIdx.x < o) red[threadIdx.x] += red[threadIdx.x+o]; __syncthreads(); }
  float sc = rsqrtf(red[0]/(float)N + 1e-8f);
  for (int j = threadIdx.x; j < N; j += 256) {
    float y = x[j]*sc*scale[j];
    Y[(size_t)row*N + j] = (_Float16)silu_(y);
  }
}

__global__ void gru_update_k(const float* __restrict__ gates, const _Float16* __restrict__ deter_m,
                             float* __restrict__ deter, _Float16* __restrict__ deter16,
                             float* __restrict__ dout)
{
  int idx = blockIdx.x*blockDim.x + threadIdx.x;
  if (idx >= NB*DET) return;
  int b = idx >> 11, d = idx & 2047;
  const float* g = gates + (size_t)b*(3*DET);
  float reset = sigm_(g[d]);
  float cand  = tanhf(reset * g[DET + d]);
  float upd   = sigm_(g[2*DET + d] - 1.0f);
  float nd = upd*cand + (1.0f - upd)*(float)deter_m[idx];
  deter[idx]   = nd;
  deter16[idx] = (_Float16)nd;
  dout[(size_t)b*NT*DET + d] = nd;   // d_seq[b][t][d], t folded into base ptr
}

__global__ void sample_k(const float* __restrict__ logits, _Float16* __restrict__ stoch,
                         float* __restrict__ sout, int t)
{
  int idx = blockIdx.x*blockDim.x + threadIdx.x;
  if (idx >= NB*32) return;
  int b = idx >> 5, s = idx & 31;
  const float* lg = logits + (size_t)b*NT*SDIM + s*32;   // post_seq[b][t][s*32..]
  float mx = -1e30f;
#pragma unroll
  for (int c = 0; c < 32; ++c) mx = fmaxf(mx, lg[c]);
  float e[32]; float sum = 0.0f;
#pragma unroll
  for (int c = 0; c < 32; ++c) { e[c] = expf(lg[c]-mx); sum += e[c]; }
  float inv = 1.0f/sum;
  int best = 0; float bestv = -1e30f;
#pragma unroll
  for (int c = 0; c < 32; ++c) {
    float p = 0.99f*e[c]*inv + 0.01f/32.0f;
    unsigned int h = (((unsigned)t*NB + (unsigned)b)*32u + (unsigned)s)*32u + (unsigned)c;
    float v = logf(p) + gumbel_hash_(h);
    if (v > bestv) { bestv = v; best = c; }
  }
#pragma unroll
  for (int c = 0; c < 32; ++c) {
    float o = (c == best) ? 1.0f : 0.0f;
    stoch[(size_t)b*SDIM + s*32 + c] = (_Float16)o;
    sout [(size_t)b*NT*SDIM + s*32 + c] = o;
  }
}

// ---------------------------------------------------------------------------

extern "C" void kernel_launch(void* const* d_in, const int* in_sizes, int n_in,
                              void* d_out, int out_size, void* d_ws, size_t ws_size,
                              hipStream_t stream)
{
  const float* tokens     = (const float*)d_in[0];
  const float* actions    = (const float*)d_in[1];
  const unsigned char* is_first = (const unsigned char*)d_in[2];
  const float* w_in_deter = (const float*)d_in[3];  const float* b_in_deter = (const float*)d_in[4];
  const float* w_in_stoch = (const float*)d_in[5];  const float* b_in_stoch = (const float*)d_in[6];
  const float* w_in_act   = (const float*)d_in[7];  const float* b_in_act   = (const float*)d_in[8];
  const float* s_n0 = (const float*)d_in[9];
  const float* s_n1 = (const float*)d_in[10];
  const float* s_n2 = (const float*)d_in[11];
  const float* w_blk_h = (const float*)d_in[12]; const float* b_blk_h = (const float*)d_in[13];
  const float* s_nh    = (const float*)d_in[14];
  const float* w_blk_g = (const float*)d_in[15]; const float* b_blk_g = (const float*)d_in[16];
  const float* w_p0 = (const float*)d_in[17]; const float* b_p0 = (const float*)d_in[18]; const float* s_p0 = (const float*)d_in[19];
  const float* w_p1 = (const float*)d_in[20]; const float* b_p1 = (const float*)d_in[21]; const float* s_p1 = (const float*)d_in[22];
  const float* w_p2 = (const float*)d_in[23]; const float* b_p2 = (const float*)d_in[24];
  const float* w_q0 = (const float*)d_in[25]; const float* b_q0 = (const float*)d_in[26]; const float* s_q0 = (const float*)d_in[27];
  const float* w_q1 = (const float*)d_in[28]; const float* b_q1 = (const float*)d_in[29];

  // ---- workspace: f32 state/raw buffers, then f16 weights + activations ----
  float* fbase = (float*)d_ws;
  size_t off = 0;
  auto aF = [&](size_t n){ float* p = fbase + off; off += n; return p; };
  float* deter = aF((size_t)NB*DET);      // recurrent carry kept in f32
  float* hraw  = aF((size_t)NB*DET);      // pre-RMS block-h output
  float* gates = aF((size_t)NB*3*DET);    // raw gates

  _Float16* hbase = (_Float16*)(fbase + off);
  size_t hoff = 0;
  auto aH = [&](size_t n){ _Float16* p = hbase + hoff; hoff += n; return p; };
  // f16 transposed weights [N][K]
  _Float16* tw_in_deter = aH((size_t)HID*DET);
  _Float16* tw_in_stoch = aH((size_t)HID*SDIM);
  _Float16* tw_in_act   = aH((size_t)HID*64);
  _Float16* tw_blk_h    = aH((size_t)NBLK*DPER*1792);
  _Float16* tw_blk_g    = aH((size_t)NBLK*768*DPER);
  _Float16* tw_q0       = aH((size_t)HID*2560);
  _Float16* tw_q1       = aH((size_t)SDIM*HID);
  _Float16* tw_p0       = aH((size_t)HID*DET);
  _Float16* tw_p1       = aH((size_t)HID*HID);
  _Float16* tw_p2       = aH((size_t)SDIM*HID);
  // f16 activations (GEMM A operands)
  _Float16* stoch    = aH((size_t)NB*SDIM);   // one-hot carry (exact in f16)
  _Float16* deter16  = aH((size_t)NB*DET);    // f16 copy of new deter
  _Float16* deter_m  = aH((size_t)NB*DET);
  _Float16* stoch_m  = aH((size_t)NB*SDIM);
  _Float16* act_n    = aH((size_t)NB*64);
  _Float16* tok16    = aH((size_t)NB*HID);
  _Float16* combined = aH((size_t)NB*1536);
  _Float16* xh       = aH((size_t)NB*DET);
  _Float16* hq       = aH((size_t)NB*HID);
  _Float16* hp       = aH((size_t)NB*HID);
  _Float16* hp2      = aH((size_t)NB*HID);

  // ---- prep: weights -> f16 [N][K] ----
  auto tr = [&](const float* src, _Float16* dst, int nblk, int K, int N){
    int total = nblk*K*N;
    transpose_f16_k<<<(total+255)/256, 256, 0, stream>>>(src, dst, nblk, K, N);
  };
  tr(w_in_deter, tw_in_deter, 1, DET,  HID);
  tr(w_in_stoch, tw_in_stoch, 1, SDIM, HID);
  tr(w_in_act,   tw_in_act,   1, 64,   HID);
  tr(w_blk_h,    tw_blk_h,    NBLK, 1792, DPER);
  tr(w_blk_g,    tw_blk_g,    NBLK, DPER, 768);
  tr(w_q0,       tw_q0,       1, 2560, HID);
  tr(w_q1,       tw_q1,       1, HID,  SDIM);
  tr(w_p0,       tw_p0,       1, DET,  HID);
  tr(w_p1,       tw_p1,       1, HID,  HID);
  tr(w_p2,       tw_p2,       1, HID,  SDIM);

  // ---- zero carries ----
  fill0f_k<<<((NB*DET)+255)/256, 256, 0, stream>>>(deter, NB*DET);
  fill0h_k<<<((NB*SDIM)+255)/256, 256, 0, stream>>>(stoch, NB*SDIM);

  float* dseq = (float*)d_out;
  float* sseq = dseq + (size_t)NB*NT*DET;
  float* pseq = sseq + (size_t)NB*NT*SDIM;
  float* qseq = pseq + (size_t)NB*NT*SDIM;

  const long long LSEQ = (long long)NT*SDIM;   // row stride in [B][T][1024] outputs

  for (int t = 0; t < NT; ++t) {
    begin_step_k<<<(NB*DET+255)/256, 256, 0, stream>>>(deter, stoch, actions, tokens,
                                                       is_first, t,
                                                       deter_m, stoch_m, act_n, tok16);
    // combined = [silu(rms(deter@Wd)) | silu(rms(stoch@Ws)) | silu(rms(act@Wa))]  (f16)
    gemm_wmma<4,1><<<dim3(4,1), 256, 0, stream>>>(deter_m, DET, DET, nullptr, 0, 0,
        tw_in_deter, b_in_deter, s_n0, combined+0,    1536, HID, DET, 0, 0);
    gemm_wmma<4,1><<<dim3(4,1), 256, 0, stream>>>(stoch_m, SDIM, SDIM, nullptr, 0, 0,
        tw_in_stoch, b_in_stoch, s_n1, combined+512,  1536, HID, SDIM, 0, 0);
    gemm_wmma<4,1><<<dim3(4,1), 256, 0, stream>>>(act_n, 64, 64, nullptr, 0, 0,
        tw_in_act,   b_in_act,   s_n2, combined+1024, 1536, HID, 64, 0, 0);
    // block h: A = [deter_blk(256) | combined(1536)], per block N=256 -> f32 raw
    gemm_wmma<2,0><<<dim3(4,NBLK), 256, 0, stream>>>(deter_m, DET, DPER, combined, 1536, 1536,
        tw_blk_h, b_blk_h, nullptr, hraw, DET, DPER, 1792, DPER, DPER);
    rmsnorm_silu_rows_k<<<NB, 256, 0, stream>>>(hraw, s_nh, xh, DET);
    // block gates: per block N=768 -> f32 raw
    gemm_wmma<6,0><<<dim3(4,NBLK), 256, 0, stream>>>(xh, DET, DPER, nullptr, 0, 0,
        tw_blk_g, b_blk_g, nullptr, gates, 3*DET, 768, DPER, DPER, 768);
    gru_update_k<<<(NB*DET+255)/256, 256, 0, stream>>>(gates, deter_m, deter, deter16,
                                                       dseq + (size_t)t*DET);
    // posterior: q0 on [new_deter | token_t], q1 -> post_seq
    gemm_wmma<4,1><<<dim3(4,1), 256, 0, stream>>>(deter16, DET, DET,
        tok16, HID, HID,
        tw_q0, b_q0, s_q0, hq, HID, HID, 2560, 0, 0);
    gemm_wmma<8,0><<<dim3(4,1), 256, 0, stream>>>(hq, HID, HID, nullptr, 0, 0,
        tw_q1, b_q1, nullptr, qseq + (size_t)t*SDIM, LSEQ, SDIM, HID, 0, 0);
    sample_k<<<(NB*32+255)/256, 256, 0, stream>>>(qseq + (size_t)t*SDIM, stoch,
                                                  sseq + (size_t)t*SDIM, t);
    // prior: p0 -> p1 -> p2 -> pri_seq
    gemm_wmma<4,1><<<dim3(4,1), 256, 0, stream>>>(deter16, DET, DET, nullptr, 0, 0,
        tw_p0, b_p0, s_p0, hp, HID, HID, DET, 0, 0);
    gemm_wmma<4,1><<<dim3(4,1), 256, 0, stream>>>(hp, HID, HID, nullptr, 0, 0,
        tw_p1, b_p1, s_p1, hp2, HID, HID, HID, 0, 0);
    gemm_wmma<8,0><<<dim3(4,1), 256, 0, stream>>>(hp2, HID, HID, nullptr, 0, 0,
        tw_p2, b_p2, nullptr, pseq + (size_t)t*SDIM, LSEQ, SDIM, HID, 0, 0);
  }
}
